// SparseMoE_4913442586647
// MI455X (gfx1250) — matrix operand order repaired and checked by reading, not compiled, not run on hardware
//
#include <hip/hip_runtime.h>
#include <math.h>

// ---------------------------------------------------------------------------
// MI455X (gfx1250) Sparse-MoE. wave32, WMMA bf16 path + async-to-LDS staging.
// N=18432 tokens, C=768, HID=1536, E=8, TOP_K=2 -> 36864 slot rows.
// Compute-bound (~2.6e11 bf16 FLOPs); x(bf16)=28MB + W(bf16)=57MB fit in 192MB L2.
// B-matrix tiles are DMA'd into LDS with GLOBAL_LOAD_ASYNC_TO_LDS_B128
// (ASYNCcnt), overlapping the v_wmma_f32_16x16x32_bf16 stream.
// ---------------------------------------------------------------------------

typedef __attribute__((ext_vector_type(16))) __bf16 v16bf;
typedef __attribute__((ext_vector_type(8)))  float  v8f;

#define NB     8
#define CDIM   768
#define HIDDIM 1536
#define NE     8
#define HWDIM  2304                 // 48*48
#define NTOK   (NB*HWDIM)           // 18432
#define NSLOT  (2*NTOK)             // 36864

// low 32 bits of a generic pointer to __shared__ = LDS byte address (ISA 10.2)
#define LDS_ADDR(p) ((unsigned)(unsigned long long)(const void*)(p))

__device__ __forceinline__ void async_load_lds16(unsigned lds_dst, const void* gsrc) {
  asm volatile("global_load_async_to_lds_b128 %0, %1, off"
               :: "v"(lds_dst), "v"((unsigned long long)gsrc)
               : "memory");
}
__device__ __forceinline__ void async_load_lds16_off2k(unsigned lds_dst, const void* gsrc) {
  // INST_OFFSET is added to BOTH the LDS and global address (ISA 08 §4.4)
  asm volatile("global_load_async_to_lds_b128 %0, %1, off offset:2048"
               :: "v"(lds_dst), "v"((unsigned long long)gsrc)
               : "memory");
}
__device__ __forceinline__ void wait_async0() {
  asm volatile("s_wait_asynccnt 0x0" ::: "memory");
}

// ------------------------- x transpose + f32->bf16 -------------------------
// in : x[b][c][hw] f32 ; out: xbf[t][c] bf16, t = b*HW + hw
__global__ __launch_bounds__(256) void convert_x_kernel(const float* __restrict__ x,
                                                        __bf16* __restrict__ xbf) {
  __shared__ float tile[32][33];
  int b  = blockIdx.z;
  int c0 = blockIdx.y * 32;
  int h0 = blockIdx.x * 32;
  int tx = threadIdx.x & 31, ty = threadIdx.x >> 5;
  for (int i = ty; i < 32; i += 8)
    tile[i][tx] = x[((size_t)(b * CDIM + c0 + i)) * HWDIM + h0 + tx];
  __syncthreads();
  for (int i = ty; i < 32; i += 8)
    xbf[((size_t)b * HWDIM + h0 + i) * CDIM + c0 + tx] = (__bf16)tile[tx][i];
}

// --------------- weight f32 -> bf16 into WMMA-B fragment blobs --------------
// Blob (e,kt,nt): 32 lanes x 32B. lane L: n = nt*16 + (L&15),
// K = kt*32 + (L>>4)*16 + {0..15} packed 2 per dword (contiguous k order, same as the A fragment).
__global__ __launch_bounds__(256) void convert_w_kernel(const float* __restrict__ w1,
                                                        const float* __restrict__ w3,
                                                        const float* __restrict__ w2,
                                                        __bf16* __restrict__ w1b,
                                                        __bf16* __restrict__ w3b,
                                                        __bf16* __restrict__ w2b) {
  int mat = blockIdx.z;                       // 0:w1 1:w3 2:w2
  const float* src; __bf16* dst; int K, Ncols;
  if (mat == 0)      { src = w1; dst = w1b; K = CDIM;   Ncols = HIDDIM; }
  else if (mat == 1) { src = w3; dst = w3b; K = CDIM;   Ncols = HIDDIM; }
  else               { src = w2; dst = w2b; K = HIDDIM; Ncols = CDIM;   }
  int NT = Ncols / 16;
  int KT = K / 32;
  int gtid = blockIdx.x * 256 + threadIdx.x;  // E*KT*NT*32 = 589824 threads
  int lane = gtid & 31;
  int blob = gtid >> 5;                       // 0 .. E*KT*NT-1 (=18432 for all mats)
  int nt = blob % NT;
  int kt = (blob / NT) % KT;
  int e  = blob / (NT * KT);
  int n     = nt * 16 + (lane & 15);
  int kbase = kt * 32 + (lane >> 4) * 16;
  v16bf v;
  #pragma unroll
  for (int i = 0; i < 16; ++i)
    v[i] = (__bf16)src[((size_t)e * K + kbase + i) * Ncols + n];
  *(v16bf*)(dst + (size_t)blob * 512 + lane * 16) = v;
}

// ------------------------------ gating ------------------------------------
__global__ __launch_bounds__(256) void gating_kernel(const float* __restrict__ x,
                                                     const float* __restrict__ noise,
                                                     const float* __restrict__ w_gate,
                                                     const float* __restrict__ b_gate,
                                                     const float* __restrict__ w_noise,
                                                     const float* __restrict__ b_noise,
                                                     int* __restrict__ slotExpert,
                                                     float* __restrict__ slotWeight) {
  __shared__ float sg[CDIM * NE];   // 24 KB
  __shared__ float sn[CDIM * NE];   // 24 KB
  for (int i = threadIdx.x; i < CDIM * NE; i += 256) { sg[i] = w_gate[i]; sn[i] = w_noise[i]; }
  __syncthreads();

  int t  = blockIdx.x * 256 + threadIdx.x;      // NTOK = 72*256 exactly
  int b  = t / HWDIM;
  int hw = t % HWDIM;
  const float* xp = x + (size_t)b * CDIM * HWDIM + hw;

  float ag[NE], an[NE];
  #pragma unroll
  for (int e = 0; e < NE; ++e) { ag[e] = 0.f; an[e] = 0.f; }
  for (int c = 0; c < CDIM; ++c) {
    float xv = xp[(size_t)c * HWDIM];
    #pragma unroll
    for (int e = 0; e < NE; ++e) {
      ag[e] += xv * sg[c * NE + e];
      an[e] += xv * sn[c * NE + e];
    }
  }
  float noisy[NE];
  #pragma unroll
  for (int e = 0; e < NE; ++e) {
    float nl = an[e] + b_noise[e];
    float sp = (nl > 30.f) ? nl : log1pf(expf(nl));   // softplus
    noisy[e] = ag[e] + b_gate[e] + noise[(size_t)t * NE + e] * sp;
  }
  // top-2 (ties -> lowest index, matching lax.top_k)
  int i0 = 0; float v0 = noisy[0];
  #pragma unroll
  for (int e = 1; e < NE; ++e) if (noisy[e] > v0) { v0 = noisy[e]; i0 = e; }
  int i1 = -1; float v1 = -3.402823466e38f;
  #pragma unroll
  for (int e = 0; e < NE; ++e) if (e != i0 && noisy[e] > v1) { v1 = noisy[e]; i1 = e; }
  float ex = expf(v1 - v0);
  float den = 1.f + ex;
  float p0 = 1.f / den;     // softmax prob of expert i0
  float p1 = ex / den;      // softmax prob of expert i1
  // reference quirk: slot_w = probs[:, :2] -> columns for EXPERTS 0 and 1
  float probs0 = (i0 == 0) ? p0 : ((i1 == 0) ? p1 : 0.f);
  float probs1 = (i0 == 1) ? p0 : ((i1 == 1) ? p1 : 0.f);
  slotExpert[2 * t]     = i0;  slotWeight[2 * t]     = probs0;
  slotExpert[2 * t + 1] = i1;  slotWeight[2 * t + 1] = probs1;
}

// ------------------------------ routing -----------------------------------
__global__ __launch_bounds__(256) void route_kernel(const int* __restrict__ slotExpert,
                                                    int* __restrict__ cnt,
                                                    int* __restrict__ rowlist) {
  int s = blockIdx.x * 256 + threadIdx.x;
  if (s >= NSLOT) return;
  int e = slotExpert[s];
  int pos = atomicAdd(&cnt[e], 1);
  rowlist[e * NSLOT + pos] = s;
}

// ----------------- GEMM1: h = silu(X@W1+b1) * (X@W3+b3) --------------------
// tile 128 rows x 64 cols, K = CDIM, 8 waves each own 16 rows.
// B tiles: async DMA global->LDS (ASYNCcnt), double buffered.
__global__ __launch_bounds__(256) void moe_gemm1_kernel(const __bf16* __restrict__ xbf,
                                                        const __bf16* __restrict__ w1b,
                                                        const __bf16* __restrict__ w3b,
                                                        const float* __restrict__ b1,
                                                        const float* __restrict__ b3,
                                                        const int* __restrict__ rowlist,
                                                        const int* __restrict__ cnt,
                                                        __bf16* __restrict__ hbuf) {
  constexpr int KT = CDIM / 32;     // 24
  constexpr int NT = HIDDIM / 16;   // 96
  int e = blockIdx.z;
  int count = cnt[e];
  int rt = blockIdx.y;
  if (rt * 128 >= count) return;    // sparse routing: skip empty tiles
  int nt0 = blockIdx.x * 4;

  int tid  = threadIdx.x;
  int wv   = tid >> 5;
  int lane = tid & 31;

  __shared__ int4 ldsB[2][2][256];  // double-buffered, 2 matrices x 4KB
  unsigned ldsBase = LDS_ADDR(&ldsB[0][0][0]);

  // gathered A row per lane (ISA A-frag: lanes&15 = M row)
  int srowA = rt * 128 + wv * 16 + (lane & 15);
  int slotA = rowlist[e * NSLOT + (srowA < count ? srowA : 0)];
  const __bf16* ap = xbf + (size_t)(slotA >> 1) * CDIM + (lane >> 4) * 16;

  int smat = tid >> 7;      // 0..1 : which matrix this half-block stages
  int sidx = tid & 127;     // 16B chunk within first 2KB of the 4KB region
  const char* wsel = smat ? (const char*)w3b : (const char*)w1b;
  auto stage = [&](int buf, int kt) {
    const char* src = wsel + ((((size_t)e * KT + kt) * NT + nt0) << 10) + sidx * 16;
    unsigned dst = ldsBase + buf * 8192 + smat * 4096 + sidx * 16;
    async_load_lds16(dst, src);          // bytes [0,2048) of region, contiguous
    async_load_lds16_off2k(dst, src);    // bytes [2048,4096), contiguous
  };

  v8f zero = {0.f,0.f,0.f,0.f,0.f,0.f,0.f,0.f};
  v8f accG[4] = {zero, zero, zero, zero};
  v8f accU[4] = {zero, zero, zero, zero};

  stage(0, 0);
  wait_async0();
  __syncthreads();
  int cur = 0;
  for (int kt = 0; kt < KT; ++kt) {
    if (kt + 1 < KT) stage(cur ^ 1, kt + 1);   // DMA next tile, overlap WMMAs
    // A fragment: lanes 0-15 K{0..15}, lanes 16-31 K{16..31}
    v16bf a;
    ((int4*)&a)[0] = *(const int4*)(ap + (size_t)kt * 32);
    ((int4*)&a)[1] = *(const int4*)(ap + (size_t)kt * 32 + 8);
    #pragma unroll
    for (int nt = 0; nt < 4; ++nt) {
      v16bf bg = *(const v16bf*)((const char*)&ldsB[cur][0][0] + nt * 1024 + lane * 32);
      accG[nt] = __builtin_amdgcn_wmma_f32_16x16x32_bf16(false, a, false, bg,
                                                         (short)0, accG[nt], false, false);
      v16bf bu = *(const v16bf*)((const char*)&ldsB[cur][1][0] + nt * 1024 + lane * 32);
      accU[nt] = __builtin_amdgcn_wmma_f32_16x16x32_bf16(false, a, false, bu,
                                                         (short)0, accU[nt], false, false);
    }
    wait_async0();                              // staging done (after compute)
    cur ^= 1;
    __syncthreads();
  }

  // epilogue: D layout -> lane row = (lane>=16?8:0)+r, col = lane&15
  int rbase = (lane >> 4) * 8;
  #pragma unroll
  for (int nt = 0; nt < 4; ++nt) {
    int col = blockIdx.x * 64 + nt * 16 + (lane & 15);
    float bb1 = b1[e * HIDDIM + col];
    float bb3 = b3[e * HIDDIM + col];
    #pragma unroll
    for (int r = 0; r < 8; ++r) {
      int srow = rt * 128 + wv * 16 + rbase + r;
      if (srow < count) {
        int slot = rowlist[e * NSLOT + srow];
        float g = accG[nt][r] + bb1;
        float u = accU[nt][r] + bb3;
        float h = (g / (1.f + expf(-g))) * u;   // silu(g) * u
        hbuf[(size_t)slot * HIDDIM + col] = (__bf16)h;
      }
    }
  }
}

// ------------- GEMM2: y += ((H@W2)+b2) * slot_weight (scatter) --------------
__global__ __launch_bounds__(256) void moe_gemm2_kernel(const __bf16* __restrict__ hbuf,
                                                        const __bf16* __restrict__ w2b,
                                                        const float* __restrict__ b2,
                                                        const int* __restrict__ rowlist,
                                                        const int* __restrict__ cnt,
                                                        const float* __restrict__ slotWeight,
                                                        float* __restrict__ ybuf) {
  constexpr int KT = HIDDIM / 32;   // 48
  constexpr int NT = CDIM / 16;     // 48
  int e = blockIdx.z;
  int count = cnt[e];
  int rt = blockIdx.y;
  if (rt * 128 >= count) return;
  int nt0 = blockIdx.x * 4;

  int tid  = threadIdx.x;
  int wv   = tid >> 5;
  int lane = tid & 31;

  __shared__ int4 ldsB[2][256];     // double-buffered 4KB
  unsigned ldsBase = LDS_ADDR(&ldsB[0][0]);

  int srowA = rt * 128 + wv * 16 + (lane & 15);
  int slotA = rowlist[e * NSLOT + (srowA < count ? srowA : 0)];
  const __bf16* ap = hbuf + (size_t)slotA * HIDDIM + (lane >> 4) * 16;

  auto stage = [&](int buf, int kt) {
    const char* src = (const char*)w2b + ((((size_t)e * KT + kt) * NT + nt0) << 10) + tid * 16;
    async_load_lds16(ldsBase + buf * 4096 + tid * 16, src);  // contiguous 4KB
  };

  v8f zero = {0.f,0.f,0.f,0.f,0.f,0.f,0.f,0.f};
  v8f acc[4] = {zero, zero, zero, zero};

  stage(0, 0);
  wait_async0();
  __syncthreads();
  int cur = 0;
  for (int kt = 0; kt < KT; ++kt) {
    if (kt + 1 < KT) stage(cur ^ 1, kt + 1);
    v16bf a;
    ((int4*)&a)[0] = *(const int4*)(ap + (size_t)kt * 32);
    ((int4*)&a)[1] = *(const int4*)(ap + (size_t)kt * 32 + 8);
    #pragma unroll
    for (int nt = 0; nt < 4; ++nt) {
      v16bf bfrag = *(const v16bf*)((const char*)&ldsB[cur][0] + nt * 1024 + lane * 32);
      acc[nt] = __builtin_amdgcn_wmma_f32_16x16x32_bf16(false, a, false, bfrag,
                                                        (short)0, acc[nt], false, false);
    }
    wait_async0();
    cur ^= 1;
    __syncthreads();
  }

  int rbase = (lane >> 4) * 8;
  #pragma unroll
  for (int nt = 0; nt < 4; ++nt) {
    int col = blockIdx.x * 64 + nt * 16 + (lane & 15);
    float bb2 = b2[e * CDIM + col];
    #pragma unroll
    for (int r = 0; r < 8; ++r) {
      int srow = rt * 128 + wv * 16 + rbase + r;
      if (srow < count) {
        int slot = rowlist[e * NSLOT + srow];
        float w  = slotWeight[slot];
        float v  = (acc[nt][r] + bb2) * w;
        // exactly 2 commutative f32 adds per y element -> deterministic
        atomicAdd(&ybuf[(size_t)(slot >> 1) * CDIM + col], v);
      }
    }
  }
}

// ---------------------- y (N,C) -> out (B,C,H,W) ---------------------------
__global__ __launch_bounds__(256) void write_out_kernel(const float* __restrict__ ybuf,
                                                        float* __restrict__ out) {
  __shared__ float tile[32][33];
  int b  = blockIdx.z;
  int c0 = blockIdx.y * 32;
  int h0 = blockIdx.x * 32;
  int tx = threadIdx.x & 31, ty = threadIdx.x >> 5;
  for (int i = ty; i < 32; i += 8)
    tile[i][tx] = ybuf[((size_t)b * HWDIM + h0 + i) * CDIM + c0 + tx];
  __syncthreads();
  for (int i = ty; i < 32; i += 8)
    out[((size_t)(b * CDIM + c0 + i)) * HWDIM + h0 + tx] = tile[tx][i];
}

// ---------------------------------------------------------------------------
extern "C" void kernel_launch(void* const* d_in, const int* in_sizes, int n_in,
                              void* d_out, int out_size, void* d_ws, size_t ws_size,
                              hipStream_t stream) {
  const float* x       = (const float*)d_in[0];
  const float* noise   = (const float*)d_in[1];
  const float* w_gate  = (const float*)d_in[2];
  const float* b_gate  = (const float*)d_in[3];
  const float* w_noise = (const float*)d_in[4];
  const float* b_noise = (const float*)d_in[5];
  const float* w1      = (const float*)d_in[6];
  const float* b1      = (const float*)d_in[7];
  const float* w2      = (const float*)d_in[8];
  const float* b2      = (const float*)d_in[9];
  const float* w3      = (const float*)d_in[10];
  const float* b3      = (const float*)d_in[11];
  float* out = (float*)d_out;

  char* base = (char*)d_ws;
  size_t off = 0;
  auto alloc = [&](size_t bytes) -> void* {
    void* r = base + off;
    off = (off + bytes + 255) & ~(size_t)255;
    return r;
  };
  __bf16* xbf  = (__bf16*)alloc((size_t)NTOK * CDIM * 2);        // 28.3 MB
  __bf16* w1b  = (__bf16*)alloc((size_t)NE * CDIM * HIDDIM * 2); // 18.9 MB
  __bf16* w3b  = (__bf16*)alloc((size_t)NE * CDIM * HIDDIM * 2);
  __bf16* w2b  = (__bf16*)alloc((size_t)NE * HIDDIM * CDIM * 2);
  __bf16* hbuf = (__bf16*)alloc((size_t)NSLOT * HIDDIM * 2);     // 113 MB
  float*  ybuf = (float*)alloc((size_t)NTOK * CDIM * 4);         // 56.6 MB
  int*    sExp = (int*)alloc((size_t)NSLOT * 4);
  float*  sWgt = (float*)alloc((size_t)NSLOT * 4);
  int*    cnt  = (int*)alloc(256);
  int*    rowl = (int*)alloc((size_t)NE * NSLOT * 4);            // 1.2 MB

  hipMemsetAsync(cnt, 0, 256, stream);
  hipMemsetAsync(ybuf, 0, (size_t)NTOK * CDIM * 4, stream);

  convert_x_kernel<<<dim3(HWDIM / 32, CDIM / 32, NB), 256, 0, stream>>>(x, xbf);
  convert_w_kernel<<<dim3(2304, 1, 3), 256, 0, stream>>>(w1, w3, w2, w1b, w3b, w2b);
  gating_kernel<<<NTOK / 256, 256, 0, stream>>>(x, noise, w_gate, b_gate, w_noise, b_noise,
                                                sExp, sWgt);
  route_kernel<<<NSLOT / 256, 256, 0, stream>>>(sExp, cnt, rowl);
  moe_gemm1_kernel<<<dim3(HIDDIM / 64, NSLOT / 128, NE), 256, 0, stream>>>(
      xbf, w1b, w3b, b1, b3, rowl, cnt, hbuf);
  moe_gemm2_kernel<<<dim3(CDIM / 64, NSLOT / 128, NE), 256, 0, stream>>>(
      hbuf, w2b, b2, rowl, cnt, sWgt, ybuf);
  write_out_kernel<<<dim3(HWDIM / 32, CDIM / 32, NB), 256, 0, stream>>>(ybuf, out);
}
